// TransformerLayerShard_69793218560668
// MI455X (gfx1250) — matrix operand order repaired and checked by reading, not compiled, not run on hardware
//
#include <hip/hip_runtime.h>

// ---------------- problem constants ----------------
#define B_   2
#define T_   2048
#define D_   2048
#define H_   16
#define Dh_  128
#define DFF_ 8192

typedef __attribute__((ext_vector_type(16))) __bf16        v16bf;
typedef __attribute__((ext_vector_type(8)))  float         v8f;
typedef __attribute__((ext_vector_type(4)))  unsigned int  u32x4;

// ---------------- WMMA fragment helpers (CDNA5 wave32 layouts) ----------------
// A-matrix 16x32 bf16: lanes 0-15 -> M=lane, elems 0..7 = K 0..7, elems 8..15 = K 16..23
//                      lanes 16-31 -> same M, K offset +8.
static __device__ __forceinline__ v16bf load_a_frag(const __bf16* base, int lda, int mBase) {
  const int lane = threadIdx.x & 31;
  const int kh   = (lane >> 4) << 3;               // 0 or 8
  const __bf16* p = base + (size_t)(mBase + (lane & 15)) * lda + kh;
  v16bf a;
#pragma unroll
  for (int i = 0; i < 8; ++i) { a[i] = p[i]; a[8 + i] = p[16 + i]; }
  return a;
}

// B-matrix 32x16 bf16, sourced from an [N][K]-layout buffer (same K order as A frag):
// lanes 0-15: elems 0..7 = K ko+0..7, elems 8..15 = K ko+16..23; lanes 16-31: K offset +8. N = nBase + (lane&15).
static __device__ __forceinline__ v16bf load_b_frag(const __bf16* bt, int ldb, int nBase, int ko) {
  const int lane = threadIdx.x & 31;
  const __bf16* p = bt + (size_t)(nBase + (lane & 15)) * ldb + ko + ((lane >> 4) << 3);
  v16bf b;
#pragma unroll
  for (int i = 0; i < 8; ++i) { b[i] = p[i]; b[8 + i] = p[16 + i]; }
  return b;
}

static __device__ __forceinline__ v8f wmma_bf16(v16bf a, v16bf b, v8f c) {
  return __builtin_amdgcn_wmma_f32_16x16x32_bf16(false, a, false, b, (short)0, c, false, false);
}

// async global -> LDS copy of 16 bytes (ASYNCcnt-tracked, no VGPR round trip).
// LDS destination address = low 32 bits of the generic pointer (aperture mapping).
static __device__ __forceinline__ void async_copy_b128(const void* gptr, void* lptr) {
  unsigned lds = (unsigned)(reinterpret_cast<uintptr_t>(lptr));
  asm volatile("global_load_async_to_lds_b128 %0, %1, off"
               :: "v"(lds), "v"(gptr) : "memory");
}
static __device__ __forceinline__ void async_wait0() {
  asm volatile("s_wait_asynccnt 0x0" ::: "memory");
}

// ---------------- f32 -> bf16 conversion ----------------
__global__ __launch_bounds__(256) void cvt_f32_bf16(const float* __restrict__ src,
                                                    __bf16* __restrict__ dst, long long n) {
  long long i = (long long)blockIdx.x * blockDim.x + threadIdx.x;
  long long stride = (long long)gridDim.x * blockDim.x;
  for (; i < n; i += stride) dst[i] = (__bf16)src[i];
}

// ---------------- LayerNorm: one block per row ----------------
__global__ __launch_bounds__(256) void ln_kernel(const float* __restrict__ x,
                                                 const float* __restrict__ g,
                                                 const float* __restrict__ bta,
                                                 __bf16* __restrict__ out) {
  __shared__ float red[256];
  const int row = blockIdx.x;
  const float* xr = x + (size_t)row * D_;
  float s = 0.f;
  for (int i = threadIdx.x; i < D_; i += 256) s += xr[i];
  red[threadIdx.x] = s;
  __syncthreads();
  for (int o = 128; o > 0; o >>= 1) {
    if (threadIdx.x < o) red[threadIdx.x] += red[threadIdx.x + o];
    __syncthreads();
  }
  const float mean = red[0] * (1.f / D_);
  __syncthreads();
  float v = 0.f;
  for (int i = threadIdx.x; i < D_; i += 256) { float d = xr[i] - mean; v += d * d; }
  red[threadIdx.x] = v;
  __syncthreads();
  for (int o = 128; o > 0; o >>= 1) {
    if (threadIdx.x < o) red[threadIdx.x] += red[threadIdx.x + o];
    __syncthreads();
  }
  const float rstd = rsqrtf(red[0] * (1.f / D_) + 1e-5f);
  __bf16* outr = out + (size_t)row * D_;
  for (int i = threadIdx.x; i < D_; i += 256)
    outr[i] = (__bf16)((xr[i] - mean) * rstd * g[i] + bta[i]);
}

// ---------------- Tiled WMMA GEMM: C[M,N] = A[M,K] * B[K,N] (+epilogue) ----------------
// Block tile 128x128, BK=32, double-buffered LDS; 8 waves as 4(M)x2(N);
// wave tile 32x64 -> 8 WMMAs per stage, async copies for stage t+1 overlap stage t math.
// EPI 0: Cbf = acc                      (QKV projections)
// EPI 1: Cf  = resid + acc              (attention out proj + residual)
// EPI 2: Cbf = gelu(acc + bias)         (MLP up)
// EPI 3: Cf  = acc + bias + resid       (MLP down + residual)
template <int EPI>
__global__ __launch_bounds__(256) void gemm_bf16_kernel(
    const __bf16* __restrict__ A, const __bf16* __restrict__ Bm,
    float* __restrict__ Cf, __bf16* __restrict__ Cbf,
    const float* __restrict__ bias, const float* __restrict__ resid,
    int M, int N, int K) {
  __shared__ __bf16 As[2][128 * 40];   // [m][k], row stride 40 (80B, 16B aligned)
  __shared__ __bf16 Bs[2][128 * 40];   // transposed: [n][k]
  const int tid  = threadIdx.x;
  const int lane = tid & 31, wave = tid >> 5;
  const int half = lane >> 4, lcol = lane & 15;
  const int mBase = blockIdx.y * 128, nBase = blockIdx.x * 128;
  const int m0 = (wave >> 1) * 32, n0 = (wave & 1) * 64;
  v8f acc[2][4] = {};

  auto stage = [&](int kBase, int buf) {
    // A tile 128x32: async global->LDS B128 copies (ASYNCcnt path, no VGPR round trip)
#pragma unroll
    for (int i = 0; i < 2; ++i) {
      int e8  = tid + i * 256;                         // 0..511 chunks of 8 bf16
      int row = e8 >> 2, koff = (e8 & 3) << 3;
      async_copy_b128(A + (size_t)(mBase + row) * K + kBase + koff,
                      &As[buf][row * 40 + koff]);
    }
    // B tile 32x128 transposed -> Bs[n][k] (vector global read, scalar LDS scatter)
#pragma unroll
    for (int i = 0; i < 2; ++i) {
      int e8 = tid + i * 256;                          // 0..511 chunks of 8 along n
      int kr = e8 >> 4, n8 = (e8 & 15) << 3;
      u32x4 val = *reinterpret_cast<const u32x4*>(Bm + (size_t)(kBase + kr) * N + nBase + n8);
      __bf16 tmp[8];
      *reinterpret_cast<u32x4*>(tmp) = val;
#pragma unroll
      for (int jj = 0; jj < 8; ++jj) Bs[buf][(n8 + jj) * 40 + kr] = tmp[jj];
    }
  };

  stage(0, 0);
  async_wait0();
  __syncthreads();

  const int nt = K >> 5;
  for (int t = 0; t < nt; ++t) {
    const int cur = t & 1;
    // fragment loads first: WMMAs below depend only on these ds_loads
    v16bf a0 = load_a_frag(As[cur], 40, m0);
    v16bf a1 = load_a_frag(As[cur], 40, m0 + 16);
    v16bf bfr[4];
#pragma unroll
    for (int ni = 0; ni < 4; ++ni) bfr[ni] = load_b_frag(Bs[cur], 40, n0 + ni * 16, 0);
    // issue next stage into the other buffer; copies overlap the WMMA burst
    if (t + 1 < nt) stage((t + 1) << 5, cur ^ 1);
#pragma unroll
    for (int ni = 0; ni < 4; ++ni) {
      acc[0][ni] = wmma_bf16(a0, bfr[ni], acc[0][ni]);
      acc[1][ni] = wmma_bf16(a1, bfr[ni], acc[1][ni]);
    }
    if (t + 1 < nt) async_wait0();
    __syncthreads();   // everyone done reading cur + writing cur^1
  }

#pragma unroll
  for (int mi = 0; mi < 2; ++mi)
#pragma unroll
    for (int ni = 0; ni < 4; ++ni)
#pragma unroll
      for (int r = 0; r < 8; ++r) {
        int row = mBase + m0 + mi * 16 + r + half * 8;   // C layout: M = r + 8*half
        int col = nBase + n0 + ni * 16 + lcol;           //           N = lane & 15
        size_t idx = (size_t)row * N + col;
        float a = acc[mi][ni][r];
        if (EPI == 0) {
          Cbf[idx] = (__bf16)a;
        } else if (EPI == 1) {
          Cf[idx] = resid[idx] + a;
        } else if (EPI == 2) {
          float t2 = a + bias[col];
          float g = 0.5f * t2 * (1.f + tanhf(0.7978845608f * (t2 + 0.044715f * t2 * t2 * t2)));
          Cbf[idx] = (__bf16)g;
        } else {
          Cf[idx] = a + bias[col] + resid[idx];
        }
      }
}

// ---------------- RoPE on q,k (first 64 dims of each head) ----------------
__global__ __launch_bounds__(256) void rope_kernel(__bf16* __restrict__ q, __bf16* __restrict__ k) {
  int idx = blockIdx.x * 256 + threadIdx.x;   // B*T*H*32 pairs
  int i = idx & 31;
  int h = (idx >> 5) & (H_ - 1);
  int t = (idx >> 9) & (T_ - 1);
  int b = idx >> 20;
  float ang = (float)t * __powf(10000.f, -(float)i * (1.f / 32.f));
  float c, s;
  __sincosf(ang, &c, &s);
  size_t base = ((size_t)(b * T_ + t)) * D_ + h * Dh_ + 2 * i;
  float q0 = (float)q[base], q1 = (float)q[base + 1];
  q[base]     = (__bf16)(q0 * c - q1 * s);
  q[base + 1] = (__bf16)(q1 * c + q0 * s);
  float k0 = (float)k[base], k1 = (float)k[base + 1];
  k[base]     = (__bf16)(k0 * c - k1 * s);
  k[base + 1] = (__bf16)(k1 * c + k0 * s);
}

// ---------------- Flash attention: 128 q-rows per block, 16 per wave ----------------
__global__ __launch_bounds__(256) void attn_kernel(const __bf16* __restrict__ q,
                                                   const __bf16* __restrict__ k,
                                                   const __bf16* __restrict__ v,
                                                   __bf16* __restrict__ ctx) {
  __shared__ __bf16 Kt[64 * 136];     // K tile, row-major [key][d]  -> S B-frags
  __shared__ __bf16 Vt[128 * 72];     // V tile, transposed [d][key] -> O B-frags
  __shared__ __bf16 Ps[8 * 16 * 72];  // per-wave P tiles [16][64] (C->A layout relay)
  const int tid  = threadIdx.x;
  const int lane = tid & 31, wave = tid >> 5;
  const int half = lane >> 4, lcol = lane & 15;
  const int qBase = blockIdx.x * 128;
  const int h = blockIdx.y, b = blockIdx.z;
  const int qr0 = qBase + wave * 16;
  const size_t headOff = (size_t)b * T_ * D_ + (size_t)h * Dh_;

  // Q fragments: 16 rows x 128 (4 K-chunks of 32)
  v16bf qa[4];
  {
    const __bf16* qp = q + headOff + (size_t)qr0 * D_;
#pragma unroll
    for (int kc = 0; kc < 4; ++kc) qa[kc] = load_a_frag(qp + kc * 32, D_, 0);
  }
  float mrow[8], lrow[8];
  v8f Oacc[8] = {};
#pragma unroll
  for (int r = 0; r < 8; ++r) { mrow[r] = -1e30f; lrow[r] = 0.f; }
  const float sc = 0.08838834764831845f;   // 1/sqrt(128)

  for (int jBase = 0; jBase <= qBase + 127; jBase += 64) {
    // stage K tile (64x128, row-major) via async global->LDS copies
#pragma unroll
    for (int i = 0; i < 4; ++i) {
      int e8 = tid + i * 256;                 // 0..1023
      int key = e8 >> 4, d = (e8 & 15) << 3;
      async_copy_b128(k + headOff + (size_t)(jBase + key) * D_ + d, &Kt[key * 136 + d]);
    }
    // stage V tile transposed (Vt[d][key]) - needs transpose, manual path
#pragma unroll
    for (int i = 0; i < 4; ++i) {
      int e8 = tid + i * 256;
      int key = e8 >> 4, d0 = (e8 & 15) << 3;
      u32x4 val = *reinterpret_cast<const u32x4*>(v + headOff + (size_t)(jBase + key) * D_ + d0);
      __bf16 tmp[8];
      *reinterpret_cast<u32x4*>(tmp) = val;
#pragma unroll
      for (int jj = 0; jj < 8; ++jj) Vt[(d0 + jj) * 72 + key] = tmp[jj];
    }
    async_wait0();
    __syncthreads();

    if (jBase <= qr0 + 15) {                  // wave-level causal tile skip
      // S = Q * K^T  (16 x 64)
      v8f sfr[4] = {};
#pragma unroll
      for (int nt = 0; nt < 4; ++nt)
#pragma unroll
        for (int kc = 0; kc < 4; ++kc)
          sfr[nt] = wmma_bf16(qa[kc], load_b_frag(Kt, 136, nt * 16, kc * 32), sfr[nt]);

      // scale + causal mask + row max
      float rmax[8];
#pragma unroll
      for (int r = 0; r < 8; ++r) rmax[r] = mrow[r];
#pragma unroll
      for (int nt = 0; nt < 4; ++nt)
#pragma unroll
        for (int r = 0; r < 8; ++r) {
          int row = qr0 + r + half * 8;
          int col = jBase + nt * 16 + lcol;
          float sv = sfr[nt][r] * sc;
          sv = (col <= row) ? sv : -1e30f;
          sfr[nt][r] = sv;
          rmax[r] = fmaxf(rmax[r], sv);
        }
#pragma unroll
      for (int r = 0; r < 8; ++r)
#pragma unroll
        for (int msk = 1; msk < 16; msk <<= 1)
          rmax[r] = fmaxf(rmax[r], __shfl_xor(rmax[r], msk, 32));

      float scaleF[8], rsum[8];
#pragma unroll
      for (int r = 0; r < 8; ++r) {
        scaleF[r] = __expf(mrow[r] - rmax[r]);
        mrow[r] = rmax[r];
        rsum[r] = 0.f;
      }
#pragma unroll
      for (int nt = 0; nt < 4; ++nt)
#pragma unroll
        for (int r = 0; r < 8; ++r) {
          float p = __expf(sfr[nt][r] - mrow[r]);
          sfr[nt][r] = p;
          rsum[r] += p;
        }
#pragma unroll
      for (int r = 0; r < 8; ++r) {
#pragma unroll
        for (int msk = 1; msk < 16; msk <<= 1)
          rsum[r] += __shfl_xor(rsum[r], msk, 32);
        lrow[r] = lrow[r] * scaleF[r] + rsum[r];
      }
#pragma unroll
      for (int ot = 0; ot < 8; ++ot)
#pragma unroll
        for (int r = 0; r < 8; ++r) Oacc[ot][r] *= scaleF[r];

      // relay P through per-wave LDS (C-layout -> A-layout)
      __bf16* P = &Ps[wave * 16 * 72];
#pragma unroll
      for (int nt = 0; nt < 4; ++nt)
#pragma unroll
        for (int r = 0; r < 8; ++r)
          P[(r + half * 8) * 72 + nt * 16 + lcol] = (__bf16)sfr[nt][r];
      // wave-coherent LDS RAW: drain DS counter before cross-lane reload
      asm volatile("s_wait_dscnt 0x0" ::: "memory");
      v16bf pa0 = load_a_frag(P, 72, 0);        // keys 0..31
      v16bf pa1 = load_a_frag(P + 32, 72, 0);   // keys 32..63
#pragma unroll
      for (int ot = 0; ot < 8; ++ot) {
        Oacc[ot] = wmma_bf16(pa0, load_b_frag(Vt, 72, ot * 16, 0),  Oacc[ot]);
        Oacc[ot] = wmma_bf16(pa1, load_b_frag(Vt, 72, ot * 16, 32), Oacc[ot]);
      }
    }
    __syncthreads();
  }

  // normalize + write ctx (bf16, feeds wo GEMM)
#pragma unroll
  for (int ot = 0; ot < 8; ++ot)
#pragma unroll
    for (int r = 0; r < 8; ++r) {
      int row = qr0 + r + half * 8;
      int d = ot * 16 + lcol;
      ctx[headOff + (size_t)row * D_ + d] = (__bf16)(Oacc[ot][r] / lrow[r]);
    }
}

// ---------------- host launcher ----------------
extern "C" void kernel_launch(void* const* d_in, const int* in_sizes, int n_in,
                              void* d_out, int out_size, void* d_ws, size_t ws_size,
                              hipStream_t stream) {
  const float* x   = (const float*)d_in[0];
  const float* lng = (const float*)d_in[1];
  const float* lnb = (const float*)d_in[2];
  const float* wq  = (const float*)d_in[3];
  const float* wk  = (const float*)d_in[4];
  const float* wv  = (const float*)d_in[5];
  const float* wo  = (const float*)d_in[6];
  const float* w1  = (const float*)d_in[7];
  const float* b1  = (const float*)d_in[8];
  const float* w2  = (const float*)d_in[9];
  const float* b2  = (const float*)d_in[10];
  float* out = (float*)d_out;

  const size_t DD  = (size_t)D_ * D_;
  const size_t DF  = (size_t)D_ * DFF_;
  const size_t BTD = (size_t)B_ * T_ * D_;
  const size_t BTF = (size_t)B_ * T_ * DFF_;

  char* ws = (char*)d_ws;
  size_t off = 0;
  auto alloc = [&](size_t bytes) -> void* {
    void* p = ws + off;
    off += (bytes + 255) & ~(size_t)255;
    return p;
  };
  __bf16* wq_b  = (__bf16*)alloc(DD * 2);
  __bf16* wk_b  = (__bf16*)alloc(DD * 2);
  __bf16* wv_b  = (__bf16*)alloc(DD * 2);
  __bf16* wo_b  = (__bf16*)alloc(DD * 2);
  __bf16* w1_b  = (__bf16*)alloc(DF * 2);
  __bf16* w2_b  = (__bf16*)alloc(DF * 2);
  __bf16* xn    = (__bf16*)alloc(BTD * 2);
  __bf16* qb    = (__bf16*)alloc(BTD * 2);
  __bf16* kb    = (__bf16*)alloc(BTD * 2);
  __bf16* vb    = (__bf16*)alloc(BTD * 2);
  __bf16* ctx   = (__bf16*)alloc(BTD * 2);
  float*  x1    = (float*) alloc(BTD * 4);
  __bf16* xn2   = (__bf16*)alloc(BTD * 2);
  __bf16* hbuf  = (__bf16*)alloc(BTF * 2);

  // 1) weights -> bf16
  cvt_f32_bf16<<<2048, 256, 0, stream>>>(wq, wq_b, (long long)DD);
  cvt_f32_bf16<<<2048, 256, 0, stream>>>(wk, wk_b, (long long)DD);
  cvt_f32_bf16<<<2048, 256, 0, stream>>>(wv, wv_b, (long long)DD);
  cvt_f32_bf16<<<2048, 256, 0, stream>>>(wo, wo_b, (long long)DD);
  cvt_f32_bf16<<<4096, 256, 0, stream>>>(w1, w1_b, (long long)DF);
  cvt_f32_bf16<<<4096, 256, 0, stream>>>(w2, w2_b, (long long)DF);

  const int M = B_ * T_;   // 4096
  // 2) LN1
  ln_kernel<<<M, 256, 0, stream>>>(x, lng, lnb, xn);
  // 3) Q,K,V projections
  dim3 gProj(D_ / 128, M / 128);
  gemm_bf16_kernel<0><<<gProj, 256, 0, stream>>>(xn, wq_b, nullptr, qb, nullptr, nullptr, M, D_, D_);
  gemm_bf16_kernel<0><<<gProj, 256, 0, stream>>>(xn, wk_b, nullptr, kb, nullptr, nullptr, M, D_, D_);
  gemm_bf16_kernel<0><<<gProj, 256, 0, stream>>>(xn, wv_b, nullptr, vb, nullptr, nullptr, M, D_, D_);
  // 4) RoPE
  rope_kernel<<<(B_ * T_ * H_ * 32) / 256, 256, 0, stream>>>(qb, kb);
  // 5) attention
  attn_kernel<<<dim3(T_ / 128, H_, B_), 256, 0, stream>>>(qb, kb, vb, ctx);
  // 6) out proj + residual -> x1 (f32)
  gemm_bf16_kernel<1><<<gProj, 256, 0, stream>>>(ctx, wo_b, x1, nullptr, nullptr, x, M, D_, D_);
  // 7) LN2
  ln_kernel<<<M, 256, 0, stream>>>(x1, lng, lnb, xn2);
  // 8) MLP up + gelu
  gemm_bf16_kernel<2><<<dim3(DFF_ / 128, M / 128), 256, 0, stream>>>(xn2, w1_b, nullptr, hbuf, b1, nullptr, M, DFF_, D_);
  // 9) MLP down + bias + residual -> out (f32)
  gemm_bf16_kernel<3><<<gProj, 256, 0, stream>>>(hbuf, w2_b, out, nullptr, b2, x1, M, D_, DFF_);
}
